// Decoder_48859547959519
// MI455X (gfx1250) — 16-bit and 8-bit matrix operand order repaired and checked by reading, compiled, not run on hardware
//
#include <hip/hip_runtime.h>


typedef __attribute__((ext_vector_type(16))) __bf16 v16bf;
typedef __attribute__((ext_vector_type(16))) int    v16i;
typedef __attribute__((ext_vector_type(8)))  float  v8f;

#define SEQ_LEN 512
#define IN_DIM  23
#define HID     256
#define FEAT    128
#define BATCH   8

#define OFF_WHH 0
#define SZ_WHH  (64 * 2 * 2048)
#define OFF_WFC (OFF_WHH + SZ_WHH)
#define SZ_WFC  (16 * 1024)
#define IMG_H8  0
#define IMG_H16 4096
#define IMG_X   12288
#define IMG_SZ  13312
#define OFF_IMG (OFF_WFC + SZ_WFC)
#define OFF_B   (OFF_IMG + 2 * IMG_SZ)
#define SZ_B    4096
#define OFF_BFC (OFF_B + SZ_B)
#define SZ_BFC  128
#define SMEM_BYTES (OFF_BFC + SZ_BFC)

#define FP8_SCALE   16.f
#define FP8_INV2    (1.f / 256.f)

__device__ __forceinline__ unsigned short f32_to_bf16(float x) {
  union { float f; unsigned u; } cv; cv.f = x;
  unsigned u = cv.u;
  u += 0x7FFFu + ((u >> 16) & 1u);
  return (unsigned short)(u >> 16);
}

__device__ __forceinline__ unsigned f32_to_e4m3(float x) {
  union { float f; unsigned u; } cv; cv.f = x;
  unsigned s = (cv.u >> 24) & 0x80u;
  float ax = fabsf(x);
  if (ax >= 448.f) return s | 0x7Eu;
  if (ax < 9.765625e-4f) return s;
  unsigned u = cv.u & 0x7FFFFFFFu;
  u += 0x0007FFFFu + ((u >> 20) & 1u);
  int e = (int)(u >> 23) - 127 + 7;
  unsigned man = (u >> 20) & 7u;
  if (e <= 0) {
    int sh = 1 - e;
    unsigned m4 = 8u | man;
    m4 = (sh > 3) ? 0u : (m4 >> sh);
    return s | m4;
  }
  if (e > 15) return s | 0x7Eu;
  return s | ((unsigned)e << 3) | man;
}

__device__ __forceinline__ float sigm(float x) { return 1.f / (1.f + __expf(-x)); }
__device__ __forceinline__ float tanh_fast(float x) { return 2.f * sigm(2.f * x) - 1.f; }

__device__ __forceinline__ int a16_off(int m, int kk) {
  int hi   = kk >> 4;
  int lane = m + (hi << 4);
  return lane * 32 + (kk & 15) * 2;
}
__device__ __forceinline__ int a8_off(int m, int k) {
  int s = k >> 7, kk = k & 127;
  int half = kk >> 6, t64 = kk & 63;
  int hi   = (t64 >> 3) & 1;
  int lane = m + (hi << 4);
  int v    = (half << 3) + ((t64 >> 4) << 1) + ((t64 & 7) >> 2);
  return s * 2048 + lane * 64 + v * 4 + (t64 & 3);
}
__device__ __forceinline__ int b8_off(int nn, int kk) {
  int hi = (kk >> 3) & 1;
  int lane = nn + (hi << 4);
  return lane * 64 + ((kk >> 6) << 5) + (((kk >> 4) & 3) << 3) + (kk & 7);
}
__device__ __forceinline__ int b16_off(int nn, int kk) {
  int hi = kk >> 4, r16 = kk & 15;
  int lane = nn + (hi << 4);
  return lane * 32 + (r16 >> 1) * 4 + ((r16 & 1) << 1);
}

__global__ __launch_bounds__(512, 1)
void lstm_decoder_persistent(const float* __restrict__ feature,
                             const float* __restrict__ W_ih,
                             const float* __restrict__ W_hh,
                             const float* __restrict__ b_ih,
                             const float* __restrict__ b_hh,
                             const float* __restrict__ W_fc,
                             const float* __restrict__ b_fc,
                             const float* __restrict__ W_hfc,
                             const float* __restrict__ b_hfc,
                             float* __restrict__ out) {
  extern __shared__ char smem[];
  const int tid  = threadIdx.x;
  const int lane = tid & 31;
  const int w    = tid >> 5;

  for (int i = tid; i < (2 * IMG_SZ) / 4; i += 512)
    ((int*)(smem + OFF_IMG))[i] = 0;
  __syncthreads();

  for (int e = tid; e < 64 * 2 * 16 * 128; e += 512) {
    int kk = e & 127, nn = (e >> 7) & 15, s = (e >> 11) & 1, T = e >> 12;
    int np = T * 16 + nn, j = np >> 2, g = np & 3;
    float wv = W_hh[(g * 256 + j) * HID + (s * 128 + kk)] * FP8_SCALE;
    smem[OFF_WHH + (T * 2 + s) * 2048 + b8_off(nn, kk)] = (char)f32_to_e4m3(wv);
  }
  for (int e = tid; e < 2 * 8 * 16 * 32; e += 512) {
    int kk = e & 31, nn = (e >> 5) & 15, ks = (e >> 9) & 7, tf = e >> 12;
    int np = tf * 16 + nn;
    float wv = (np < IN_DIM) ? W_fc[np * HID + (ks * 32 + kk)] : 0.f;
    *(unsigned short*)(smem + OFF_WFC + (tf * 8 + ks) * 1024 + b16_off(nn, kk)) =
        f32_to_bf16(wv);
  }
  for (int n = tid; n < 1024; n += 512) {
    int j = n >> 2, g = n & 3;
    ((float*)(smem + OFF_B))[n] = b_ih[g * 256 + j] + b_hh[g * 256 + j];
  }
  if (tid < 32) ((float*)(smem + OFF_BFC))[tid] = (tid < IN_DIM) ? b_fc[tid] : 0.f;
  for (int e = tid; e < BATCH * HID; e += 512) {
    int m = e >> 8, j = e & 255;
    float acc = b_hfc[j];
    for (int f = 0; f < FEAT; ++f) acc += feature[m * FEAT + f] * W_hfc[j * FEAT + f];
    smem[OFF_IMG + IMG_H8 + a8_off(m, j)] = (char)f32_to_e4m3(acc * FP8_SCALE);
    *(unsigned short*)(smem + OFF_IMG + IMG_H16 + (j >> 5) * 1024 + a16_off(m, j & 31)) =
        f32_to_bf16(acc);
  }
  union UB { unsigned u[8]; v16bf v; };
  UB wih[4];
#pragma unroll
  for (int tt = 0; tt < 4; ++tt) {
    int T = w * 4 + tt, nn = lane & 15, hi = lane >> 4;
#pragma unroll
    for (int v = 0; v < 8; ++v) {
      unsigned pk = 0;
#pragma unroll
      for (int h = 0; h < 2; ++h) {
        int kk = 2 * v + h + 16 * hi;
        int np = T * 16 + nn, j = np >> 2, g = np & 3;
        float wv = (kk < IN_DIM) ? W_ih[(g * 256 + j) * IN_DIM + kk] : 0.f;
        pk |= ((unsigned)f32_to_bf16(wv)) << (16 * h);
      }
      wih[tt].u[v] = pk;
    }
  }
  float bn[4];
#pragma unroll
  for (int tt = 0; tt < 4; ++tt)
    bn[tt] = ((const float*)(smem + OFF_B))[(w * 4 + tt) * 16 + (lane & 15)];
  const float bfc = ((const float*)(smem + OFF_BFC))[(w & 1) * 16 + (lane & 15)];
  float creg[4][8];
#pragma unroll
  for (int tt = 0; tt < 4; ++tt)
#pragma unroll
    for (int r = 0; r < 8; ++r) creg[tt][r] = 0.f;
  __syncthreads();

#pragma unroll 1
  for (int t = 0; t < SEQ_LEN; ++t) {
    const char* rb = smem + OFF_IMG + (t & 1) * IMG_SZ;
    char*       wb = smem + OFF_IMG + ((t & 1) ^ 1) * IMG_SZ;

    v16i  a8lo = *(const v16i*)(rb + IMG_H8 + lane * 64);
    v16i  a8hi = *(const v16i*)(rb + IMG_H8 + 2048 + lane * 64);
    v16bf ax   = *(const v16bf*)(rb + IMG_X + lane * 32);

    v8f gacc[4];
#pragma unroll
    for (int tt = 0; tt < 4; ++tt) {
      int T = w * 4 + tt;
      v8f accB;
#pragma unroll
      for (int r = 0; r < 8; ++r) accB[r] = 0.f;
      accB = __builtin_amdgcn_wmma_f32_16x16x32_bf16(
          false, ax, false, wih[tt].v, (short)0, accB, false, false);
      v8f acc8;
#pragma unroll
      for (int r = 0; r < 8; ++r) acc8[r] = 0.f;
      v16i b0 = *(const v16i*)(smem + OFF_WHH + (T * 2 + 0) * 2048 + lane * 64);
      v16i b1 = *(const v16i*)(smem + OFF_WHH + (T * 2 + 1) * 2048 + lane * 64);
      acc8 = __builtin_amdgcn_wmma_f32_16x16x128_fp8_fp8(a8lo, b0, (short)0, acc8, false, false);
      acc8 = __builtin_amdgcn_wmma_f32_16x16x128_fp8_fp8(a8hi, b1, (short)0, acc8, false, false);
#pragma unroll
      for (int r = 0; r < 8; ++r)
        gacc[tt][r] = (accB[r] + bn[tt]) + acc8[r] * FP8_INV2;
    }

    if (lane < 16) {
      const int q0 = ((lane & ~3) + 0) << 2, q1 = ((lane & ~3) + 1) << 2;
      const int q2 = ((lane & ~3) + 2) << 2, q3 = ((lane & ~3) + 3) << 2;
#pragma unroll
      for (int tt = 0; tt < 4; ++tt) {
#pragma unroll
        for (int r = 0; r < 8; ++r) {
          int gv = __float_as_int(gacc[tt][r]);
          float iv = __int_as_float(__builtin_amdgcn_ds_bpermute(q0, gv));
          float fv = __int_as_float(__builtin_amdgcn_ds_bpermute(q1, gv));
          float gg = __int_as_float(__builtin_amdgcn_ds_bpermute(q2, gv));
          float ov = __int_as_float(__builtin_amdgcn_ds_bpermute(q3, gv));
          float cn = sigm(fv) * creg[tt][r] + sigm(iv) * tanh_fast(gg);
          float hn = sigm(ov) * tanh_fast(cn);
          creg[tt][r] = cn;
          if ((lane & 3) == 0) {
            int j = (w * 4 + tt) * 4 + (lane >> 2);
            wb[IMG_H8 + a8_off(r, j)] = (char)f32_to_e4m3(hn * FP8_SCALE);
            *(unsigned short*)(wb + IMG_H16 + (j >> 5) * 1024 + a16_off(r, j & 31)) =
                f32_to_bf16(hn);
          }
        }
      }
    }
    __syncthreads();

    if (w < 2) {
      v8f accY;
#pragma unroll
      for (int r = 0; r < 8; ++r) accY[r] = 0.f;
#pragma unroll
      for (int ks = 0; ks < 8; ++ks) {
        v16bf av = *(const v16bf*)(wb + IMG_H16 + ks * 1024 + lane * 32);
        v16bf bv = *(const v16bf*)(smem + OFF_WFC + (w * 8 + ks) * 1024 + lane * 32);
        accY = __builtin_amdgcn_wmma_f32_16x16x32_bf16(
            false, av, false, bv, (short)0, accY, false, false);
      }
      if (lane < 16) {
        int n = w * 16 + lane;
        if (n < IN_DIM) {
#pragma unroll
          for (int r = 0; r < 8; ++r) {
            float y = accY[r] + bfc;
            out[(r * SEQ_LEN + t) * IN_DIM + n] = y;
            *(unsigned short*)(wb + IMG_X + a16_off(r, n)) = f32_to_bf16(y);
          }
        }
      }
    }
    __syncthreads();
  }
}

extern "C" void kernel_launch(void* const* d_in, const int* in_sizes, int n_in,
                              void* d_out, int out_size, void* d_ws, size_t ws_size,
                              hipStream_t stream) {
  (void)in_sizes; (void)n_in; (void)out_size; (void)d_ws; (void)ws_size;
  const float* feature = (const float*)d_in[0];
  const float* W_ih    = (const float*)d_in[1];
  const float* W_hh    = (const float*)d_in[2];
  const float* b_ih    = (const float*)d_in[3];
  const float* b_hh    = (const float*)d_in[4];
  const float* W_fc    = (const float*)d_in[5];
  const float* b_fc    = (const float*)d_in[6];
  const float* W_hfc   = (const float*)d_in[7];
  const float* b_hfc   = (const float*)d_in[8];
  float* out = (float*)d_out;

  (void)hipFuncSetAttribute((const void*)lstm_decoder_persistent,
                            hipFuncAttributeMaxDynamicSharedMemorySize, SMEM_BYTES);
  lstm_decoder_persistent<<<1, 512, SMEM_BYTES, stream>>>(
      feature, W_ih, W_hh, b_ih, b_hh, W_fc, b_fc, W_hfc, b_hfc, out);
}
